// GatedDeltaNet_2018634629556
// MI455X (gfx1250) — matrix operand order repaired and checked by reading, not compiled, not run on hardware
//
#include <hip/hip_runtime.h>
#include <hip/hip_bf16.h>
#include <math.h>

// ---------------- problem dims (compile-time) ----------------
#define Bq   2
#define Tq   2048
#define HID  2048
#define HK   16
#define HV   32
#define DK   128
#define DV   128
#define KW   4
#define KEY_DIM   (HK * DK)               // 2048
#define VAL_DIM   (HV * DV)               // 4096
#define CONV_DIM  (2 * KEY_DIM + VAL_DIM) // 8192
#define EPSq 1e-6f

typedef __attribute__((ext_vector_type(16))) __bf16 v16bf;
typedef __attribute__((ext_vector_type(8)))  float  v8f;

// load 8 consecutive fp32, convert to bf16 into fragment slots [base, base+8)
__device__ __forceinline__ void load8_bf16(const float* __restrict__ p,
                                           v16bf& f, int base) {
  const float4 u = *(const float4*)p;
  const float4 v = *(const float4*)(p + 4);
  f[base + 0] = (__bf16)u.x; f[base + 1] = (__bf16)u.y;
  f[base + 2] = (__bf16)u.z; f[base + 3] = (__bf16)u.w;
  f[base + 4] = (__bf16)v.x; f[base + 5] = (__bf16)v.y;
  f[base + 6] = (__bf16)v.z; f[base + 7] = (__bf16)v.w;
}

// =====================================================================
// WMMA GEMM:  C[M,N] = A[M,K] * W[N,K]^T   (fp32 mem, bf16 MACs, f32 acc)
// Wave tile 64x48 = 4 M-tiles x 3 N-tiles = 12 WMMA per 32-K step.
// acc = 96 VGPRs, A frags 32, B frags 24 -> stays well under the 256-VGPR
// cliff so all K-step loads get independent destinations (clause-batched).
// Branch-free inner loop: OOB tiles clamp their load rows; stores guarded.
// 4 waves / block -> block tile 64 x 192.  Requires K % 32 == 0, N % 16 == 0.
// =====================================================================
__global__ void __launch_bounds__(128)
wmma_gemm_bf16(const float* __restrict__ A, const float* __restrict__ W,
               float* __restrict__ C, int M, int N, int K) {
  const int wave = threadIdx.x >> 5;
  const int lane = threadIdx.x & 31;
  const int hlf  = lane >> 4;        // lane half (0/1)
  const int l16  = lane & 15;

  const int m0    = blockIdx.y * 64;                   // 4 M-tiles
  const int nbase = (blockIdx.x * 12 + wave * 3) * 16; // 3 N-tiles
  if (m0 >= M || nbase >= N) return;                   // wave-uniform exit

  // Load-row bases, clamped into range so the K-loop is branch-free.
  const float* Arow[4];
  const float* Wrow[3];
  bool mok[4], nok[3];
#pragma unroll
  for (int i = 0; i < 4; ++i) {
    const int mt = m0 + i * 16;
    mok[i] = (mt < M);
    Arow[i] = A + (size_t)(min(mt, M - 16) + l16) * K;   // A: lanes carry M rows
  }
#pragma unroll
  for (int j = 0; j < 3; ++j) {
    const int nt = nbase + j * 16;
    nok[j] = (nt < N);
    Wrow[j] = W + (size_t)(min(nt, N - 16) + l16) * K;   // B: lanes carry N rows
  }

  v8f acc[4][3];
#pragma unroll
  for (int i = 0; i < 4; ++i)
#pragma unroll
    for (int j = 0; j < 3; ++j) acc[i][j] = (v8f){};

  for (int k0 = 0; k0 < K; k0 += 32) {
    // prefetch next K-block of the streamed operands into cache
    if (k0 + 32 < K) {
#pragma unroll
      for (int j = 0; j < 3; ++j) __builtin_prefetch(Wrow[j] + k0 + 32, 0, 1);
#pragma unroll
      for (int i = 0; i < 4; ++i) __builtin_prefetch(Arow[i] + k0 + 32, 0, 1);
    }

    // A 16x32 bf16 fragments: two contiguous 8-float runs per lane
    v16bf af[4];
#pragma unroll
    for (int i = 0; i < 4; ++i) {
      load8_bf16(Arow[i] + k0 + hlf * 8,      af[i], 0);
      load8_bf16(Arow[i] + k0 + 16 + hlf * 8, af[i], 8);
    }
    // B 32x16 bf16 fragments in separate buffers (loads batch into clauses)
    v16bf bf[3];
#pragma unroll
    for (int j = 0; j < 3; ++j) {
      load8_bf16(Wrow[j] + k0 + hlf * 8,      bf[j], 0);
      load8_bf16(Wrow[j] + k0 + 16 + hlf * 8, bf[j], 8);
    }
    // 12 straight-line WMMAs
#pragma unroll
    for (int j = 0; j < 3; ++j)
#pragma unroll
      for (int i = 0; i < 4; ++i)
        acc[i][j] = __builtin_amdgcn_wmma_f32_16x16x32_bf16(
            false, af[i], false, bf[j], (short)0, acc[i][j], false, false);
  }

  // C/D layout: VGPR r -> M = mt + r + 8*hlf ; N = nt + l16
#pragma unroll
  for (int i = 0; i < 4; ++i) {
    if (mok[i]) {
      const int mrow = m0 + i * 16 + hlf * 8;
#pragma unroll
      for (int j = 0; j < 3; ++j) {
        if (nok[j]) {
          const int col = nbase + j * 16 + l16;
#pragma unroll
          for (int r = 0; r < 8; ++r)
            C[(size_t)(mrow + r) * N + col] = acc[i][j][r];
        }
      }
    }
  }
}

// =====================================================================
// causal depthwise conv1d (KW=4, left pad 3) + SiLU, over [B,T,C]
// =====================================================================
__global__ void conv_silu_kernel(const float* __restrict__ mixed,
                                 const float* __restrict__ conv_w,
                                 float* __restrict__ qkv) {
  const size_t total = (size_t)Bq * Tq * CONV_DIM;
  size_t idx = (size_t)blockIdx.x * blockDim.x + threadIdx.x;
  if (idx >= total) return;
  const int c = (int)(idx % CONV_DIM);
  const size_t bt = idx / CONV_DIM;
  const int t = (int)(bt % Tq);
  const int b = (int)(bt / Tq);
  const float* w = conv_w + (size_t)c * KW;
  float acc = 0.f;
#pragma unroll
  for (int j = 0; j < KW; ++j) {
    const int tt = t - (KW - 1) + j;
    if (tt >= 0) acc += w[j] * mixed[((size_t)b * Tq + tt) * CONV_DIM + c];
  }
  qkv[idx] = acc / (1.f + __expf(-acc));   // silu
}

// =====================================================================
// L2-normalize q and k heads: one wave per (b,t,head,{q|k}) row of 128
// =====================================================================
__global__ void l2norm_qk(const float* __restrict__ qkv,
                          float* __restrict__ qn,
                          float* __restrict__ kn) {
  const int lane = threadIdx.x & 31;
  long wid = ((long)blockIdx.x * blockDim.x + threadIdx.x) >> 5;
  const int h = (int)(wid % HK);  wid /= HK;
  const int which = (int)(wid & 1); wid >>= 1;       // 0 = q, 1 = k
  const int t = (int)(wid % Tq);
  const int b = (int)(wid / Tq);

  const size_t src = ((size_t)b * Tq + t) * CONV_DIM + which * KEY_DIM + (size_t)h * DK;
  float v[4];
  float ss = 0.f;
#pragma unroll
  for (int j = 0; j < 4; ++j) {
    v[j] = qkv[src + lane * 4 + j];
    ss += v[j] * v[j];
  }
#pragma unroll
  for (int m = 16; m >= 1; m >>= 1) ss += __shfl_xor(ss, m, 32);
  const float inv = 1.f / fmaxf(sqrtf(ss), 1e-12f);

  float* dst = (which == 0 ? qn : kn);
  const size_t d = (((size_t)b * Tq + t) * HK + h) * DK;
#pragma unroll
  for (int j = 0; j < 4; ++j) dst[d + lane * 4 + j] = v[j] * inv;
}

// =====================================================================
// beta = sigmoid(braw); g = -exp(A_log) * softplus(araw + dt_bias)
// =====================================================================
__global__ void beta_g_kernel(const float* __restrict__ braw,
                              const float* __restrict__ araw,
                              const float* __restrict__ dt_bias,
                              const float* __restrict__ A_log,
                              float* __restrict__ beta,
                              float* __restrict__ g) {
  const int total = Bq * Tq * HV;
  const int i = blockIdx.x * blockDim.x + threadIdx.x;
  if (i >= total) return;
  const int hv = i % HV;
  beta[i] = 1.f / (1.f + __expf(-braw[i]));
  const float a = araw[i] + dt_bias[hv];
  const float sp = (a > 20.f) ? a : log1pf(__expf(a));
  g[i] = -__expf(A_log[hv]) * sp;
}

// =====================================================================
// gated delta rule scan: one block per (b, hv); S[128][128] f32 split
// 64 regs/thread; two LDS reduction points per step.
// =====================================================================
__global__ void __launch_bounds__(256)
gdn_scan(const float* __restrict__ qn, const float* __restrict__ kn,
         const float* __restrict__ qkv, const float* __restrict__ g,
         const float* __restrict__ beta, float* __restrict__ o) {
  const int bh = blockIdx.x;
  const int b  = bh / HV;
  const int hv = bh % HV;
  const int hk = hv >> 1;               // rep = HV/HK = 2
  const int tid   = threadIdx.x;
  const int col   = tid & 127;          // v index
  const int rhalf = tid >> 7;           // rows [rhalf*64, rhalf*64+64)

  __shared__ float sh_q[DK], sh_k[DK], sh_v[DV];
  __shared__ float sh_kv[2][DV];
  __shared__ float sh_o[2][DV];

  float S[64];
#pragma unroll
  for (int i = 0; i < 64; ++i) S[i] = 0.f;

  const float scale = 0.08838834764831845f;  // DK^-0.5

  for (int t = 0; t < Tq; ++t) {
    const size_t qk_base = (((size_t)b * Tq + t) * HK + hk) * DK;
    const size_t v_base  = ((size_t)b * Tq + t) * CONV_DIM + 2 * KEY_DIM + (size_t)hv * DV;
    if (tid < 128) {
      sh_q[tid] = qn[qk_base + tid];
      sh_k[tid] = kn[qk_base + tid];
    } else {
      sh_v[tid - 128] = qkv[v_base + (tid - 128)];
    }
    __syncthreads();

    const size_t gb = ((size_t)b * Tq + t) * HV + hv;
    const float eg = __expf(g[gb]);
    const float bt = beta[gb];

    float kv = 0.f;
#pragma unroll
    for (int i = 0; i < 64; ++i) {
      S[i] *= eg;
      kv += sh_k[rhalf * 64 + i] * S[i];
    }
    sh_kv[rhalf][col] = kv;
    __syncthreads();

    const float u = bt * (sh_v[col] - (sh_kv[0][col] + sh_kv[1][col]));
    float op = 0.f;
#pragma unroll
    for (int i = 0; i < 64; ++i) {
      S[i] += sh_k[rhalf * 64 + i] * u;
      op += sh_q[rhalf * 64 + i] * S[i];
    }
    sh_o[rhalf][col] = op;
    __syncthreads();

    if (tid < 128)
      o[(((size_t)b * Tq + t) * HV + hv) * DV + tid] =
          scale * (sh_o[0][tid] + sh_o[1][tid]);
    __syncthreads();
  }
}

// =====================================================================
// gated RMSNorm: RMSNorm(o) * weight * silu(z); one wave per (b,t,h)
// =====================================================================
__global__ void gated_norm(const float* __restrict__ o,
                           const float* __restrict__ z,
                           const float* __restrict__ w,
                           float* __restrict__ gated) {
  const int lane = threadIdx.x & 31;
  const long wid = ((long)blockIdx.x * blockDim.x + threadIdx.x) >> 5; // (b,t,h) flat
  const size_t base = (size_t)wid * DV;
  float v[4];
  float ss = 0.f;
#pragma unroll
  for (int j = 0; j < 4; ++j) {
    v[j] = o[base + lane * 4 + j];
    ss += v[j] * v[j];
  }
#pragma unroll
  for (int m = 16; m >= 1; m >>= 1) ss += __shfl_xor(ss, m, 32);
  const float inv = rsqrtf(ss / (float)DV + EPSq);
#pragma unroll
  for (int j = 0; j < 4; ++j) {
    const int d = lane * 4 + j;
    const float zz = z[base + d];
    const float sz = zz / (1.f + __expf(-zz));
    gated[base + d] = w[d] * v[j] * inv * sz;
  }
}

// =====================================================================
// host-side orchestration
// =====================================================================
extern "C" void kernel_launch(void* const* d_in, const int* in_sizes, int n_in,
                              void* d_out, int out_size, void* d_ws, size_t ws_size,
                              hipStream_t stream) {
  (void)in_sizes; (void)n_in; (void)out_size; (void)ws_size;
  const float* x        = (const float*)d_in[0];
  // d_in[1] = input_pos (unused: states start at zero)
  const float* W_qkv    = (const float*)d_in[2];
  const float* W_z      = (const float*)d_in[3];
  const float* W_b      = (const float*)d_in[4];
  const float* W_a      = (const float*)d_in[5];
  const float* conv_w   = (const float*)d_in[6];
  const float* dt_bias  = (const float*)d_in[7];
  const float* A_log    = (const float*)d_in[8];
  const float* norm_w   = (const float*)d_in[9];
  const float* W_out    = (const float*)d_in[10];
  float* out = (float*)d_out;

  const int M = Bq * Tq;  // 4096

  // workspace layout (floats); mixed_qkv region is recycled for o + gated
  float* ws = (float*)d_ws;
  const size_t n_mixed = (size_t)M * CONV_DIM;   // 33,554,432
  const size_t n_z     = (size_t)M * VAL_DIM;    // 16,777,216
  const size_t n_bh    = (size_t)M * HV;         // 131,072
  const size_t n_qk    = (size_t)M * HK * DK;    // 8,388,608

  float* mixed = ws;                        size_t off = n_mixed;
  float* qkv   = ws + off;                  off += n_mixed;
  float* zbuf  = ws + off;                  off += n_z;
  float* braw  = ws + off;                  off += n_bh;
  float* araw  = ws + off;                  off += n_bh;
  float* betab = ws + off;                  off += n_bh;
  float* gbuf  = ws + off;                  off += n_bh;
  float* qn    = ws + off;                  off += n_qk;
  float* kn    = ws + off;                  off += n_qk;
  float* obuf  = mixed;                     // recycle mixed region
  float* gated = mixed + n_z;

  dim3 blk(128);
  // block tile = 64 (M) x 192 (N); grids: x = ceil(N/192), y = M/64
  wmma_gemm_bf16<<<dim3((CONV_DIM + 191) / 192, M / 64), blk, 0, stream>>>(
      x, W_qkv, mixed, M, CONV_DIM, HID);
  wmma_gemm_bf16<<<dim3((VAL_DIM + 191) / 192, M / 64), blk, 0, stream>>>(
      x, W_z, zbuf, M, VAL_DIM, HID);
  wmma_gemm_bf16<<<dim3(1, M / 64), blk, 0, stream>>>(
      x, W_b, braw, M, HV, HID);
  wmma_gemm_bf16<<<dim3(1, M / 64), blk, 0, stream>>>(
      x, W_a, araw, M, HV, HID);

  // 2) causal depthwise conv + silu
  {
    const size_t total = (size_t)M * CONV_DIM;
    conv_silu_kernel<<<dim3((unsigned)((total + 255) / 256)), dim3(256), 0, stream>>>(
        mixed, conv_w, qkv);
  }

  // 3) l2norm q/k (one wave per row); waves = B*T*2*HK
  {
    const long waves = (long)Bq * Tq * 2 * HK;
    l2norm_qk<<<dim3((unsigned)(waves / 8)), dim3(256), 0, stream>>>(qkv, qn, kn);
  }

  // 4) beta / g gates
  beta_g_kernel<<<dim3((unsigned)((n_bh + 255) / 256)), dim3(256), 0, stream>>>(
      braw, araw, dt_bias, A_log, betab, gbuf);

  // 5) gated delta-rule scan: 64 blocks (B*HV), 256 threads each
  gdn_scan<<<dim3(Bq * HV), dim3(256), 0, stream>>>(qn, kn, qkv, gbuf, betab, obuf);

  // 6) gated RMSNorm; waves = B*T*HV
  {
    const long waves = (long)Bq * Tq * HV;
    gated_norm<<<dim3((unsigned)(waves / 8)), dim3(256), 0, stream>>>(
        obuf, zbuf, norm_w, gated);
  }

  // 7) output projection
  wmma_gemm_bf16<<<dim3((HID + 191) / 192, M / 64), blk, 0, stream>>>(
      gated, W_out, out, M, HID, VAL_DIM);
}
